// PresidentialAttention_57174604645017
// MI455X (gfx1250) — matrix operand order repaired and checked by reading, not compiled, not run on hardware
//
#include <hip/hip_runtime.h>
#include <hip/hip_bf16.h>

// ---------------------------------------------------------------------------
// MHA forward for B=4, S=2048, H=1024, NH=16, HD=64 on gfx1250 (MI455X).
// All matrix math through v_wmma_f32_16x16x32_f16 (f16 in, f32 accumulate).
// f16 tiles staged into LDS via GLOBAL_LOAD_ASYNC_TO_LDS_B128 when the
// toolchain exposes the builtin (ASYNCcnt path); fallback = VGPR staging.
// ---------------------------------------------------------------------------

constexpr int B_  = 4;
constexpr int S_  = 2048;
constexpr int H_  = 1024;
constexpr int NH_ = 16;
constexpr int HD_ = 64;
constexpr int BS_ = B_ * S_;     // 8192 rows

typedef __attribute__((ext_vector_type(16))) _Float16 v16h;
typedef __attribute__((ext_vector_type(8)))  _Float16 v8h;
typedef __attribute__((ext_vector_type(4)))  _Float16 v4h;
typedef __attribute__((ext_vector_type(8)))  float    v8f;
typedef __attribute__((ext_vector_type(4)))  float    v4f;
typedef __attribute__((ext_vector_type(4)))  int      v4i;

#if defined(__gfx1250__) && \
    __has_builtin(__builtin_amdgcn_global_load_async_to_lds_b128) && \
    __has_builtin(__builtin_amdgcn_s_wait_asynccnt)
#define USE_ASYNC_LDS 1
// builtin signature (from hipcc diagnostic): param0 = v4i addrspace(1)*,
// param1 = v4i addrspace(3)*, then imm offset + imm cpol.
typedef __attribute__((address_space(1))) v4i* gv4i_p;
typedef __attribute__((address_space(3))) v4i* lv4i_p;
#else
#define USE_ASYNC_LDS 0
#endif

union AFrag { v16h v; v8h h[2]; };

__device__ __forceinline__ v16h make16(v8h lo, v8h hi) {
    AFrag u; u.h[0] = lo; u.h[1] = hi; return u.v;
}

// ---------------------------------------------------------------------------
// GEMM: out[M,N] = A[M,K] * W[K,N] + bias[N]
//   AT = float (convert to f16 in LDS) or _Float16 (already f16 -> async copy)
//   HEADED: write f16 output in [B, NH, S, HD] layout (for Q/K/V)
//           else write f32 flat [M, N] (final projection into d_out)
// Block: 256 threads (8 wave32), tile 128x128, K stepped by 32.
// Wave w covers rows (w>>1)*32..+32, cols (w&1)*64..+64 -> 2x4 WMMA tiles.
// ---------------------------------------------------------------------------
template <typename AT, bool HEADED>
__global__ __launch_bounds__(256)
void gemm_kernel(const AT* __restrict__ A, const float* __restrict__ W,
                 const float* __restrict__ bias, void* __restrict__ outP)
{
    constexpr int K = H_;
    constexpr int N = H_;
    __shared__ __align__(16) _Float16 As[128][32];   // [m][k]
    __shared__ __align__(16) _Float16 Bs[128][32];   // [n][k] (transposed)

    const int t    = threadIdx.x;
    const int lane = t & 31;
    const int wid  = t >> 5;
    const int m0   = blockIdx.y * 128;
    const int n0   = blockIdx.x * 128;
    const int wm   = (wid >> 1) * 32;   // 0,32,64,96
    const int wn   = (wid & 1) * 64;    // 0,64

    v8f acc[2][4];
    const v8f vzero = {};
    for (int i = 0; i < 2; ++i)
        for (int j = 0; j < 4; ++j) acc[i][j] = vzero;

    const int ml    = lane & 15;
    const int koff  = (lane & 16) ? 16 : 0;   // A-frag: K half per half-wave
    const int kbase = (lane & 16) ? 16 : 0;   // B-frag: K half per half-wave

    for (int k0 = 0; k0 < K; k0 += 32) {
        // ---- stage A tile (128x32) ---------------------------------------
        if constexpr (sizeof(AT) == 2) {
#if USE_ASYNC_LDS
            // f16 source: async DMA straight into LDS (ASYNCcnt path)
            #pragma unroll
            for (int i = 0; i < 2; ++i) {
                int s   = t + 256 * i;
                int row = s >> 2;
                int cs  = (s & 3) * 8;                  // 8 halves = 16 B
                __builtin_amdgcn_global_load_async_to_lds_b128(
                    (gv4i_p)(A + (size_t)(m0 + row) * K + k0 + cs),
                    (lv4i_p)&As[row][cs], 0, 0);
            }
#else
            #pragma unroll
            for (int i = 0; i < 4; ++i) {
                int s   = t + 256 * i;
                int row = s >> 3;
                int cs  = (s & 7) * 4;
                *(v4h*)&As[row][cs] =
                    *(const v4h*)(A + (size_t)(m0 + row) * K + k0 + cs);
            }
#endif
        } else {
            // fp32 source: load, convert f32->f16, store to LDS
            #pragma unroll
            for (int i = 0; i < 4; ++i) {
                int s   = t + 256 * i;
                int row = s >> 3;
                int cs  = (s & 7) * 4;
                v4f av = *(const v4f*)(A + (size_t)(m0 + row) * K + k0 + cs);
                v4h hv;
                hv[0] = (_Float16)av[0]; hv[1] = (_Float16)av[1];
                hv[2] = (_Float16)av[2]; hv[3] = (_Float16)av[3];
                *(v4h*)&As[row][cs] = hv;
            }
        }
        // ---- stage W tile (32x128) transposed into [n][k] -----------------
        #pragma unroll
        for (int i = 0; i < 4; ++i) {
            int s  = t + 256 * i;
            int kk = s >> 5;
            int ns = (s & 31) * 4;
            v4f wv = *(const v4f*)(W + (size_t)(k0 + kk) * N + n0 + ns);
            Bs[ns + 0][kk] = (_Float16)wv[0];
            Bs[ns + 1][kk] = (_Float16)wv[1];
            Bs[ns + 2][kk] = (_Float16)wv[2];
            Bs[ns + 3][kk] = (_Float16)wv[3];
        }
        if (k0 + 32 < K) {  // pull next K-tiles toward L2 (global_prefetch_b8)
            __builtin_prefetch(A + (size_t)(m0 + (t >> 1)) * K + (k0 + 32) + (t & 1) * 16, 0, 3);
            __builtin_prefetch(W + (size_t)(k0 + 32 + (t >> 4)) * N + n0 + (t & 15) * 8, 0, 3);
        }
#if USE_ASYNC_LDS
        if constexpr (sizeof(AT) == 2)
            __builtin_amdgcn_s_wait_asynccnt(0);
#endif
        __syncthreads();

        // ---- fragments: contiguous k = 16h + s, same for A and B ------------------
        v16h afr[2], bfr[4];
        #pragma unroll
        for (int i = 0; i < 2; ++i) {
            int r = wm + i * 16 + ml;                       // M = lane%16
            afr[i] = make16(*(const v8h*)&As[r][koff],
                            *(const v8h*)&As[r][koff + 8]);
        }
        #pragma unroll
        for (int j = 0; j < 4; ++j) {
            int c = wn + j * 16 + ml;                       // N = lane%16
            bfr[j] = make16(*(const v8h*)&Bs[c][kbase],
                            *(const v8h*)&Bs[c][kbase + 8]);
        }
        #pragma unroll
        for (int i = 0; i < 2; ++i)
            #pragma unroll
            for (int j = 0; j < 4; ++j)
                acc[i][j] = __builtin_amdgcn_wmma_f32_16x16x32_f16(
                    false, afr[i], false, bfr[j], (short)0, acc[i][j], false, false);
        __syncthreads();
    }

    // ---- epilogue: bias add + store (C layout: M = r + 8*(lane>=16)) ------
    const int mb = (lane & 16) ? 8 : 0;
    #pragma unroll
    for (int i = 0; i < 2; ++i) {
        #pragma unroll
        for (int j = 0; j < 4; ++j) {
            int n    = n0 + wn + j * 16 + ml;
            float bv = bias[n];
            #pragma unroll
            for (int r = 0; r < 8; ++r) {
                int   m   = m0 + wm + i * 16 + mb + r;
                float val = acc[i][j][r] + bv;
                if constexpr (HEADED) {
                    int b = m >> 11, sI = m & (S_ - 1);
                    int h = n >> 6,  d  = n & (HD_ - 1);
                    ((_Float16*)outP)[(((size_t)b * NH_ + h) * S_ + sI) * HD_ + d] =
                        (_Float16)val;
                } else {
                    ((float*)outP)[(size_t)m * N + n] = val;
                }
            }
        }
    }
}

// ---------------------------------------------------------------------------
// Flash attention: per block 128 q rows of one (b,h); 8 waves x 16 rows.
// KV tiles of 64; online softmax; all matmuls via WMMA f16.
// Q and K tiles staged via async-to-LDS (f16, no conversion); V needs a
// transpose so it stays on the VGPR path (overlaps with the async copies).
// ---------------------------------------------------------------------------
__global__ __launch_bounds__(256)
void attn_kernel(const _Float16* __restrict__ Q, const _Float16* __restrict__ Kt,
                 const _Float16* __restrict__ V, const float* __restrict__ mask,
                 _Float16* __restrict__ O)
{
    __shared__ __align__(16) _Float16 Qs[128][HD_];   // [q][d]
    __shared__ __align__(16) _Float16 Ks[64][HD_];    // [kv][d]
    __shared__ __align__(16) _Float16 Vs[HD_][64];    // [d][kv] (transposed)
    __shared__ __align__(16) _Float16 Ps[128][64];    // P tile, per-wave 16-row slice

    const int t    = threadIdx.x;
    const int lane = t & 31;
    const int wid  = t >> 5;
    const int q0   = blockIdx.x * 128;
    const int h    = blockIdx.y;
    const int b    = blockIdx.z;

    const size_t headOff = ((size_t)b * NH_ + h) * S_ * HD_;
    const _Float16* Qp = Q  + headOff;
    const _Float16* Kp = Kt + headOff;
    const _Float16* Vp = V  + headOff;

    // ---- load Q tile once (128x64 halves) ---------------------------------
#if USE_ASYNC_LDS
    #pragma unroll
    for (int i = 0; i < 4; ++i) {
        int s = t + 256 * i;
        int row = s >> 3, cs = (s & 7) * 8;
        __builtin_amdgcn_global_load_async_to_lds_b128(
            (gv4i_p)(Qp + (size_t)(q0 + row) * HD_ + cs),
            (lv4i_p)&Qs[row][cs], 0, 0);
    }
#else
    #pragma unroll
    for (int i = 0; i < 8; ++i) {
        int s = t + 256 * i;
        int row = s >> 4, cs = (s & 15) * 4;
        *(v4h*)&Qs[row][cs] = *(const v4h*)(Qp + (size_t)(q0 + row) * HD_ + cs);
    }
#endif

    const int ml    = lane & 15;
    const int hi    = (lane & 16) ? 1 : 0;
    const int koff  = hi ? 16 : 0;
    const int kbase = hi ? 16 : 0;
    const int wq    = wid * 16;

    float mrow[8], lrow[8];
    v8f o[4];
    const v8f vzero = {};
    #pragma unroll
    for (int r = 0; r < 8; ++r) { mrow[r] = -1e30f; lrow[r] = 0.0f; }
    #pragma unroll
    for (int j = 0; j < 4; ++j) o[j] = vzero;

    constexpr float scale = 0.125f;   // 1/sqrt(64)

    for (int kv0 = 0; kv0 < S_; kv0 += 64) {
        // ---- stage K tile (async when available) --------------------------
#if USE_ASYNC_LDS
        #pragma unroll
        for (int i = 0; i < 2; ++i) {
            int s = t + 256 * i;
            int row = s >> 3, cs = (s & 7) * 8;
            __builtin_amdgcn_global_load_async_to_lds_b128(
                (gv4i_p)(Kp + (size_t)(kv0 + row) * HD_ + cs),
                (lv4i_p)&Ks[row][cs], 0, 0);
        }
#else
        #pragma unroll
        for (int i = 0; i < 4; ++i) {
            int s = t + 256 * i;
            int row = s >> 4, cs = (s & 15) * 4;
            *(v4h*)&Ks[row][cs] = *(const v4h*)(Kp + (size_t)(kv0 + row) * HD_ + cs);
        }
#endif
        // ---- stage V tile transposed [d][kv] (VGPR path) ------------------
        #pragma unroll
        for (int i = 0; i < 4; ++i) {
            int s = t + 256 * i;
            int row = s >> 4, cs = (s & 15) * 4;
            v4h vv = *(const v4h*)(Vp + (size_t)(kv0 + row) * HD_ + cs);
            Vs[cs + 0][row] = vv[0]; Vs[cs + 1][row] = vv[1];
            Vs[cs + 2][row] = vv[2]; Vs[cs + 3][row] = vv[3];
        }
#if USE_ASYNC_LDS
        __builtin_amdgcn_s_wait_asynccnt(0);
#endif
        __syncthreads();

        // ---- S = Q * K^T : 4 kv-subtiles x 2 k-steps ----------------------
        v8f sacc[4];
        #pragma unroll
        for (int j = 0; j < 4; ++j) sacc[j] = vzero;
        #pragma unroll
        for (int kk = 0; kk < 2; ++kk) {
            v16h af = make16(*(const v8h*)&Qs[wq + ml][kk * 32 + koff],
                             *(const v8h*)&Qs[wq + ml][kk * 32 + koff + 8]);
            #pragma unroll
            for (int j = 0; j < 4; ++j) {
                v16h bf = make16(*(const v8h*)&Ks[j * 16 + ml][kk * 32 + kbase],
                                 *(const v8h*)&Ks[j * 16 + ml][kk * 32 + kbase + 8]);
                sacc[j] = __builtin_amdgcn_wmma_f32_16x16x32_f16(
                    false, af, false, bf, (short)0, sacc[j], false, false);
            }
        }

        // ---- mask + online softmax (rows live on fixed VGPR across 16 lanes)
        const float* mp = mask + ((size_t)b * S_ + q0 + wq) * S_ + kv0;
        #pragma unroll
        for (int r = 0; r < 8; ++r) {
            int qr = r + 8 * hi;
            float rowmax = mrow[r];
            #pragma unroll
            for (int j = 0; j < 4; ++j) {
                float sc = sacc[j][r] * scale
                         + mp[(size_t)qr * S_ + j * 16 + ml] * (-1e9f);
                sacc[j][r] = sc;
                rowmax = fmaxf(rowmax, sc);
            }
            #pragma unroll
            for (int mk = 1; mk < 16; mk <<= 1)
                rowmax = fmaxf(rowmax, __shfl_xor(rowmax, mk, 32));
            float alpha = __expf(mrow[r] - rowmax);
            float rsum  = 0.0f;
            #pragma unroll
            for (int j = 0; j < 4; ++j) {
                float e = __expf(sacc[j][r] - rowmax);
                sacc[j][r] = e;
                rsum += e;
            }
            #pragma unroll
            for (int mk = 1; mk < 16; mk <<= 1)
                rsum += __shfl_xor(rsum, mk, 32);
            lrow[r] = lrow[r] * alpha + rsum;
            mrow[r] = rowmax;
            #pragma unroll
            for (int j = 0; j < 4; ++j) {
                o[j][r] *= alpha;
                Ps[wq + qr][j * 16 + ml] = (_Float16)sacc[j][r];
            }
        }

        // ---- O += P * V : 4 d-subtiles x 2 k-steps (LDS in-order per wave)
        #pragma unroll
        for (int kk = 0; kk < 2; ++kk) {
            v16h af = make16(*(const v8h*)&Ps[wq + ml][kk * 32 + koff],
                             *(const v8h*)&Ps[wq + ml][kk * 32 + koff + 8]);
            #pragma unroll
            for (int j = 0; j < 4; ++j) {
                v16h bf = make16(*(const v8h*)&Vs[j * 16 + ml][kk * 32 + kbase],
                                 *(const v8h*)&Vs[j * 16 + ml][kk * 32 + kbase + 8]);
                o[j] = __builtin_amdgcn_wmma_f32_16x16x32_f16(
                    false, af, false, bf, (short)0, o[j], false, false);
            }
        }
        __syncthreads();
    }

    // ---- normalize and write O in [B, S, H] f16 layout --------------------
    #pragma unroll
    for (int r = 0; r < 8; ++r) {
        float inv = 1.0f / lrow[r];
        int   qr  = q0 + wq + r + 8 * hi;
        size_t base = ((size_t)b * S_ + qr) * H_ + (size_t)h * HD_;
        #pragma unroll
        for (int j = 0; j < 4; ++j)
            O[base + j * 16 + ml] = (_Float16)(o[j][r] * inv);
    }
}

// ---------------------------------------------------------------------------
// Launch: 3 headed projections -> flash attention -> output projection.
// ---------------------------------------------------------------------------
extern "C" void kernel_launch(void* const* d_in, const int* in_sizes, int n_in,
                              void* d_out, int out_size, void* d_ws, size_t ws_size,
                              hipStream_t stream) {
    (void)in_sizes; (void)n_in; (void)out_size; (void)ws_size;
    const float* query = (const float*)d_in[0];
    const float* key   = (const float*)d_in[1];
    const float* value = (const float*)d_in[2];
    const float* mask  = (const float*)d_in[3];
    const float* Wq    = (const float*)d_in[4];
    const float* bq    = (const float*)d_in[5];
    const float* Wk    = (const float*)d_in[6];
    const float* bk    = (const float*)d_in[7];
    const float* Wv    = (const float*)d_in[8];
    const float* bv    = (const float*)d_in[9];
    const float* Wo    = (const float*)d_in[10];
    const float* bo    = (const float*)d_in[11];

    _Float16* ws  = (_Float16*)d_ws;
    const size_t per = (size_t)BS_ * H_;           // 8 Mi elements (16.78 MB f16)
    _Float16* Qws = ws;
    _Float16* Kws = ws + per;
    _Float16* Vws = ws + 2 * per;
    _Float16* Ows = ws + 3 * per;

    dim3 gGemm(H_ / 128, BS_ / 128);               // (8, 64)
    dim3 blk(256);

    gemm_kernel<float, true><<<gGemm, blk, 0, stream>>>(query, Wq, bq, (void*)Qws);
    gemm_kernel<float, true><<<gGemm, blk, 0, stream>>>(key,   Wk, bk, (void*)Kws);
    gemm_kernel<float, true><<<gGemm, blk, 0, stream>>>(value, Wv, bv, (void*)Vws);

    dim3 gAttn(S_ / 128, NH_, B_);                 // (16, 16, 4)
    attn_kernel<<<gAttn, blk, 0, stream>>>(Qws, Kws, Vws, mask, Ows);

    gemm_kernel<_Float16, false><<<gGemm, blk, 0, stream>>>(Ows, Wo, bo, d_out);
}
